// DiagonalMaskedAttention_50852412784725
// MI455X (gfx1250) — compile-verified, matrix operand order checked by reading, not run on hardware
//
#include <hip/hip_runtime.h>
#include <hip/hip_bf16.h>

typedef _Float16 half16 __attribute__((ext_vector_type(16)));
typedef _Float16 half8  __attribute__((ext_vector_type(8)));
typedef float    float8 __attribute__((ext_vector_type(8)));

#define EMBED 1024
#define HEADS 16
#define HDIM  64
#define SEQ   2048
#define BATCH 2
#define MTOT  (BATCH * SEQ)          // 4096 rows for the big GEMMs
#define ATTN_SCALE 0.125f            // 64^-0.5
#define NEG_BIG (-1.0e30f)

// ---- GEMM LDS tile geometry: 128 rows x 32 halfs, padded to 40 halfs ------
#define TILE_ROWS  128
#define LDS_PITCH  40                          // halfs (80 B pitch)
#define TILE_HALFS (TILE_ROWS * LDS_PITCH)     // 5120
#define TILE_BYTES (TILE_HALFS * 2)            // 10240
#define GEMM_SMEM  (4 * TILE_BYTES)            // A0,A1,B0,B1 = 40 KiB

// ---- Attention LDS geometry ----------------------------------------------
// K tile: 32 keys x 64 halfs, padded to 72 halfs (144 B pitch).
// V tile: 64 d-rows x 32 halfs, padded to 40 halfs (80 B pitch).
#define KT_PITCH   72
#define KT_HALFS   (32 * KT_PITCH)             // 2304 (4608 B)
#define VT_PITCH   40
#define VT_HALFS   (64 * VT_PITCH)             // 2560 (5120 B)
#define ATTN_KT_OFF(buf)  ((buf) * KT_HALFS)                 // halfs
#define ATTN_VT_OFF(buf)  (2 * KT_HALFS + (buf) * VT_HALFS)  // halfs
#define ATTN_PBUF_HOFF    (2 * KT_HALFS + 2 * VT_HALFS)      // 9728 halfs
#define ATTN_SMEM  (ATTN_PBUF_HOFF * 2 + 4 * 16 * 32 * 4)    // 27648 B

#if defined(__has_builtin)
#  if __has_builtin(__builtin_amdgcn_tensor_load_to_lds) && \
      __has_builtin(__builtin_amdgcn_s_wait_tensorcnt)
#    define HAVE_TDM 1
#  endif
#endif
#ifndef HAVE_TDM
#  define HAVE_TDM 0
#endif

#if HAVE_TDM
typedef unsigned int v4u __attribute__((ext_vector_type(4)));
typedef int          v8i __attribute__((ext_vector_type(8)));
typedef int          v4i __attribute__((ext_vector_type(4)));

// Issue one 2D TDM tile load (global->LDS). pad_icode selects the LDS row
// padding interval (3 = every 16 DWORDs, 4 = every 32 DWORDs); pad amount is
// fixed at 4 DWORDs (16 B) to de-conflict LDS banks for b128 fragment reads.
// Toolchain uses the 6-arg builtin: (g0 v4u, g1 v8i, v4i, v4i, v8i, cpol).
__device__ __forceinline__ void tdm_load_2d(const _Float16* gtile,
                                            unsigned lds_byte_addr,
                                            int tensor_d0, int tensor_d1,
                                            int tile_d0, int tile_d1,
                                            int stride0_elems, int pad_icode) {
    unsigned long long ga = (unsigned long long)(uintptr_t)gtile;
    v4u g0;
    g0.x = 1u;                                       // count=1, user descriptor
    g0.y = lds_byte_addr;                            // lds_addr
    g0.z = (unsigned)(ga & 0xFFFFFFFFu);             // global_addr[31:0]
    g0.w = (unsigned)((ga >> 32) & 0x01FFFFFFu)      // global_addr[56:32]
         | 0x80000000u;                              // type=2 ("image")
    v8i g1;
    g1[0] = (1 << 16)              // data_size = 2 bytes
          | (1 << 20)              // pad_enable
          | (pad_icode << 22)      // pad_interval (DWORDs per LDS row)
          | (3 << 25);             // pad_amount: 4 DWORDs (16 B)
    g1[1] = (tensor_d0 & 0xFFFF) << 16;                              // dim0 lo
    g1[2] = ((tensor_d0 >> 16) & 0xFFFF) | ((tensor_d1 & 0xFFFF) << 16);
    g1[3] = ((tensor_d1 >> 16) & 0xFFFF) | ((tile_d0 & 0xFFFF) << 16);
    g1[4] = (tile_d1 & 0xFFFF);                                      // tile_dim1
    g1[5] = stride0_elems;                                           // stride0 lo
    g1[6] = 0;
    g1[7] = 0;
    v4i z4 = {0, 0, 0, 0};
    v8i z8 = {0, 0, 0, 0, 0, 0, 0, 0};
    __builtin_amdgcn_tensor_load_to_lds(g0, g1, z4, z4, z8, 0);
}
#endif

// ---------------------------------------------------------------------------
// 16x32 f16 fragment from row-major global memory (A/B operand layout,
// ISA 7.12.2): lane L -> row base+L%16, K-set {0-7,16-23} / {8-15,24-31}.
// ---------------------------------------------------------------------------
__device__ __forceinline__ half16 load_frag16x32(const _Float16* __restrict__ base,
                                                 int rowBase, int ld, int kBase) {
    const int lane = threadIdx.x & 31;
    const int l16  = lane & 15;
    const int lh   = lane >> 4;
    const _Float16* p = base + (size_t)(rowBase + l16) * (size_t)ld + kBase + lh * 8;
    half8 lo = *(const half8*)(p);
    half8 hi = *(const half8*)(p + 16);
    return __builtin_shufflevector(lo, hi, 0,1,2,3,4,5,6,7,8,9,10,11,12,13,14,15);
}

// Same fragment pattern from a padded LDS tile (pitch in halfs).
__device__ __forceinline__ half16 lds_frag16x32(const _Float16* tile, int rowBase,
                                                int kBase, int pitch) {
    const int lane = threadIdx.x & 31;
    const int l16  = lane & 15;
    const int lh   = lane >> 4;
    const _Float16* p = tile + (size_t)(rowBase + l16) * pitch + kBase + lh * 8;
    half8 lo = *(const half8*)(p);
    half8 hi = *(const half8*)(p + 16);
    return __builtin_shufflevector(lo, hi, 0,1,2,3,4,5,6,7,8,9,10,11,12,13,14,15);
}

// ---------------------------------------------------------------------------
// f32 -> f16 conversion (one-shot; everything stays L2 resident afterwards)
// ---------------------------------------------------------------------------
__global__ void cvt_f32_to_f16_kernel(const float* __restrict__ in,
                                      _Float16* __restrict__ out, int n) {
    int i = blockIdx.x * blockDim.x + threadIdx.x;
    if (i < n) out[i] = (_Float16)in[i];
}

// ---------------------------------------------------------------------------
// WMMA GEMM:  C[M,N] = A[M,K] * W[N,K]^T + bias[N]
// Block: 128 thr (4 waves, 2x2). Wave computes a 64x64 C tile (16 WMMAs per
// 32-wide k-step). A/B tiles staged to LDS by the Tensor Data Mover, double
// buffered; split s_wait_tensorcnt overlaps DMA with the WMMA phase.
//   mode 0: f16 out, scatter to Q/K layout [B,H,S,Dh]
//   mode 1: f16 out, scatter to V^T layout [B,H,Dh,S]
//   mode 2: f32 out, plain row-major [M,N]
// ---------------------------------------------------------------------------
__global__ __launch_bounds__(128)
void gemm_wmma_bias_kernel(const _Float16* __restrict__ A,
                           const _Float16* __restrict__ W,
                           const float* __restrict__ bias,
                           void* __restrict__ outp,
                           int M, int N, int K, int mode) {
    extern __shared__ _Float16 smemh[];   // [A0 | A1 | B0 | B1], 40 KiB
    _Float16* Abuf0 = smemh;
    _Float16* Abuf1 = smemh + TILE_HALFS;
    _Float16* Bbuf0 = smemh + 2 * TILE_HALFS;
    _Float16* Bbuf1 = smemh + 3 * TILE_HALFS;

    const int wave = threadIdx.x >> 5;
    const int lane = threadIdx.x & 31;
    const int l16  = lane & 15;
    const int lh   = lane >> 4;
    const int mi   = wave >> 1;           // 0..1
    const int ni   = wave & 1;            // 0..1
    const int mBlock = blockIdx.x * 128;
    const int nBlock = blockIdx.y * 128;

    float8 acc[4][4];
    const float8 fzero = {0.f,0.f,0.f,0.f,0.f,0.f,0.f,0.f};
#pragma unroll
    for (int f = 0; f < 4; ++f)
#pragma unroll
        for (int g = 0; g < 4; ++g) acc[f][g] = fzero;

    const int KT = K / 32;

#if HAVE_TDM
    // Each wave DMAs its own 32-row slice of the 128x32 A and B tiles.
    auto issue = [&](int buf, int k) {
        const unsigned ldsA = (unsigned)(buf * TILE_BYTES) + wave * 32 * (LDS_PITCH * 2);
        const unsigned ldsB = (unsigned)((2 + buf) * TILE_BYTES) + wave * 32 * (LDS_PITCH * 2);
        tdm_load_2d(A + (size_t)(mBlock + wave * 32) * K + k, ldsA, K, M, 32, 32, K, 3);
        tdm_load_2d(W + (size_t)(nBlock + wave * 32) * K + k, ldsB, K, N, 32, 32, K, 3);
    };
    issue(0, 0);
    for (int kt = 0; kt < KT; ++kt) {
        const int cur = kt & 1;
        if (kt + 1 < KT) {
            issue(cur ^ 1, (kt + 1) * 32);
            __builtin_amdgcn_s_wait_tensorcnt(2);  // wait for current pair only
        } else {
            __builtin_amdgcn_s_wait_tensorcnt(0);
        }
        __syncthreads();                            // publish tiles block-wide
        const _Float16* Ab = cur ? Abuf1 : Abuf0;
        const _Float16* Bb = cur ? Bbuf1 : Bbuf0;
        half16 af[4];
#pragma unroll
        for (int f = 0; f < 4; ++f)
            af[f] = lds_frag16x32(Ab, mi * 64 + f * 16, 0, LDS_PITCH);
#pragma unroll
        for (int g = 0; g < 4; ++g) {
            half16 bf = lds_frag16x32(Bb, ni * 64 + g * 16, 0, LDS_PITCH);
#pragma unroll
            for (int f = 0; f < 4; ++f)
                acc[f][g] = __builtin_amdgcn_wmma_f32_16x16x32_f16(
                    false, af[f], false, bf, (short)0, acc[f][g], false, false);
        }
        __syncthreads();   // all reads done before next issue overwrites buf
    }
#else
    // Fallback: cooperative global->LDS staging with plain vector loads.
    for (int kt = 0; kt < KT; ++kt) {
        const int k = kt * 32;
        __syncthreads();
        {   // thread t copies row t (64 B) of each tile
            const int t = threadIdx.x;
            const half8* sa = (const half8*)(A + (size_t)(mBlock + t) * K + k);
            const half8* sb = (const half8*)(W + (size_t)(nBlock + t) * K + k);
            half8* da = (half8*)(Abuf0 + (size_t)t * LDS_PITCH);
            half8* db = (half8*)(Bbuf0 + (size_t)t * LDS_PITCH);
#pragma unroll
            for (int c = 0; c < 4; ++c) { da[c] = sa[c]; db[c] = sb[c]; }
        }
        __syncthreads();
        half16 af[4];
#pragma unroll
        for (int f = 0; f < 4; ++f)
            af[f] = lds_frag16x32(Abuf0, mi * 64 + f * 16, 0, LDS_PITCH);
#pragma unroll
        for (int g = 0; g < 4; ++g) {
            half16 bf = lds_frag16x32(Bbuf0, ni * 64 + g * 16, 0, LDS_PITCH);
#pragma unroll
            for (int f = 0; f < 4; ++f)
                acc[f][g] = __builtin_amdgcn_wmma_f32_16x16x32_f16(
                    false, af[f], false, bf, (short)0, acc[f][g], false, false);
        }
    }
#endif

    // Epilogue: C layout -> lane(0-15) rows v, lane(16-31) rows v+8.
    const int mWave = mBlock + mi * 64;
    const int nWave = nBlock + ni * 64;
#pragma unroll
    for (int f = 0; f < 4; ++f) {
#pragma unroll
        for (int g = 0; g < 4; ++g) {
            const int col = nWave + g * 16 + l16;
            const float bv = bias[col];
#pragma unroll
            for (int v = 0; v < 8; ++v) {
                const int m = mWave + f * 16 + lh * 8 + v;
                const float val = acc[f][g][v] + bv;
                if (mode == 2) {
                    ((float*)outp)[(size_t)m * N + col] = val;
                } else {
                    const int b = m >> 11;          // m / SEQ
                    const int s = m & (SEQ - 1);
                    const int h = col >> 6;         // col / HDIM
                    const int d = col & (HDIM - 1);
                    _Float16* O = (_Float16*)outp;
                    if (mode == 0) {
                        O[(((size_t)(b * HEADS + h) * SEQ + s) * HDIM) + d] = (_Float16)val;
                    } else { // mode 1: V transposed [B,H,Dh,S]
                        O[(((size_t)(b * HEADS + h) * HDIM + d) * SEQ) + s] = (_Float16)val;
                    }
                }
            }
        }
    }
}

// ---------------------------------------------------------------------------
// Flash attention: grid (S/64, B*H), block 128 (4 waves).
// Wave owns a 16-row query tile. Per 32-key step, the K (32x64) and V^T
// (64x32) tiles are TDM-staged to LDS (wave0 -> K, wave1 -> V), double
// buffered with a split s_wait_tensorcnt. All waves loop over the block's
// causal key range; a wave-uniform guard skips compute past a wave's range.
// ---------------------------------------------------------------------------
__global__ __launch_bounds__(128)
void flash_attn_wmma_kernel(const _Float16* __restrict__ Q,
                            const _Float16* __restrict__ Kd,
                            const _Float16* __restrict__ Vt,
                            const int* __restrict__ amask,
                            _Float16* __restrict__ Ctx) {
    extern __shared__ _Float16 smemh[];
    float* pbufW = (float*)(smemh + ATTN_PBUF_HOFF) + (threadIdx.x >> 5) * 16 * 32;

    const int wave = threadIdx.x >> 5;
    const int lane = threadIdx.x & 31;
    const int l16  = lane & 15;
    const int lh   = lane >> 4;
    const int bh   = blockIdx.y;        // b*HEADS + h
    const int b    = bh >> 4;
    const int h    = bh & 15;
    const int qBase = blockIdx.x * 64 + wave * 16;

    const _Float16* Qb = Q  + (size_t)bh * SEQ * HDIM;
    const _Float16* Kb = Kd + (size_t)bh * SEQ * HDIM;
    const _Float16* Vb = Vt + (size_t)bh * HDIM * SEQ;

    const half16 aq0 = load_frag16x32(Qb, qBase, HDIM, 0);
    const half16 aq1 = load_frag16x32(Qb, qBase, HDIM, 32);

    const float8 fzero = {0.f,0.f,0.f,0.f,0.f,0.f,0.f,0.f};
    float8 o[4];
#pragma unroll
    for (int nt = 0; nt < 4; ++nt) o[nt] = fzero;
    float mrow[8], lrow[8];
#pragma unroll
    for (int v = 0; v < 8; ++v) { mrow[v] = NEG_BIG; lrow[v] = 0.f; }

    const int NSTEP = blockIdx.x * 2 + 2;     // (blockIdx.x*64 + 64) / 32

#if HAVE_TDM
    auto issue = [&](int buf, int j) {
        if (wave == 0) {
            // K tile: 32 key-rows x 64 halfs, pitch 144 B (pad every 32 DW)
            tdm_load_2d(Kb + (size_t)j * HDIM, (unsigned)(ATTN_KT_OFF(buf) * 2),
                        HDIM, SEQ, HDIM, 32, HDIM, 4);
        } else if (wave == 1) {
            // V tile: 64 d-rows x 32 halfs, pitch 80 B (pad every 16 DW)
            tdm_load_2d(Vb + j, (unsigned)(ATTN_VT_OFF(buf) * 2),
                        SEQ, HDIM, 32, HDIM, SEQ, 3);
        }
    };
    issue(0, 0);
#endif

    for (int step = 0; step < NSTEP; ++step) {
        const int j   = step * 32;
        const int cur = step & 1;
#if HAVE_TDM
        if (step + 1 < NSTEP) {
            issue(cur ^ 1, j + 32);
            __builtin_amdgcn_s_wait_tensorcnt(1);   // our current tile only
        } else {
            __builtin_amdgcn_s_wait_tensorcnt(0);
        }
        __syncthreads();                             // publish K/V tiles
        const _Float16* Kt = smemh + ATTN_KT_OFF(cur);
        const _Float16* Vtile = smemh + ATTN_VT_OFF(cur);
#else
        __syncthreads();
        {   // cooperative staging: 128 threads, K row t/4 chunk t%4, V row t/2 chunk t%2
            const int t = threadIdx.x;
            const half8* sk = (const half8*)(Kb + (size_t)(j + (t >> 2)) * HDIM) + (t & 3);
            half8* dk = (half8*)(smemh + ATTN_KT_OFF(cur) + (size_t)(t >> 2) * KT_PITCH) + (t & 3);
            dk[0] = sk[0];
            const half8* sv = (const half8*)(Vb + (size_t)(t >> 1) * SEQ + j) + (t & 1);
            half8* dv = (half8*)(smemh + ATTN_VT_OFF(cur) + (size_t)(t >> 1) * VT_PITCH) + (t & 1);
            dv[0] = sv[0];
        }
        __syncthreads();
        const _Float16* Kt = smemh + ATTN_KT_OFF(cur);
        const _Float16* Vtile = smemh + ATTN_VT_OFF(cur);
#endif

        if (j < qBase + 16) {   // wave-uniform causal guard (EXEC stays all-1)
            // ---- scores for 32 keys: two 16x16 WMMA outputs ---------------
            float8 c[2];
#pragma unroll
            for (int sub = 0; sub < 2; ++sub) {
                half16 kb0 = lds_frag16x32(Kt, sub * 16, 0,  KT_PITCH);
                half16 kb1 = lds_frag16x32(Kt, sub * 16, 32, KT_PITCH);
                float8 z = fzero;
                z = __builtin_amdgcn_wmma_f32_16x16x32_f16(false, aq0, false, kb0,
                                                           (short)0, z, false, false);
                z = __builtin_amdgcn_wmma_f32_16x16x32_f16(false, aq1, false, kb1,
                                                           (short)0, z, false, false);
                c[sub] = z;
            }

            // ---- scale + causal + key-padding mask ------------------------
            const int kc0 = j + l16;
            const int kc1 = j + 16 + l16;
            const bool pm0 = amask[b * SEQ + kc0] != 0;
            const bool pm1 = amask[b * SEQ + kc1] != 0;
#pragma unroll
            for (int v = 0; v < 8; ++v) {
                const int qr = qBase + lh * 8 + v;
                float s0 = c[0][v] * ATTN_SCALE;
                float s1 = c[1][v] * ATTN_SCALE;
                if (kc0 > qr || !pm0) s0 = NEG_BIG;
                if (kc1 > qr || !pm1) s1 = NEG_BIG;
                c[0][v] = s0; c[1][v] = s1;
            }

            // ---- online softmax (row reductions across 16-lane half) ------
            float corr[8];
#pragma unroll
            for (int v = 0; v < 8; ++v) {
                float pm = fmaxf(c[0][v], c[1][v]);
#pragma unroll
                for (int d = 1; d < 16; d <<= 1) pm = fmaxf(pm, __shfl_xor(pm, d, 32));
                const float nm = fmaxf(mrow[v], pm);
                const float cr = __expf(mrow[v] - nm);
                const float p0 = __expf(c[0][v] - nm);
                const float p1 = __expf(c[1][v] - nm);
                c[0][v] = p0; c[1][v] = p1;
                float rs = p0 + p1;
#pragma unroll
                for (int d = 1; d < 16; d <<= 1) rs += __shfl_xor(rs, d, 32);
                lrow[v] = lrow[v] * cr + rs;
                mrow[v] = nm;
                corr[v] = cr;
            }
#pragma unroll
            for (int nt = 0; nt < 4; ++nt)
#pragma unroll
                for (int v = 0; v < 8; ++v) o[nt][v] *= corr[v];

            // ---- C-layout -> A-layout for P via per-wave LDS tile ---------
#pragma unroll
            for (int v = 0; v < 8; ++v) {
                pbufW[(lh * 8 + v) * 32 + l16]      = c[0][v];
                pbufW[(lh * 8 + v) * 32 + 16 + l16] = c[1][v];
            }
            asm volatile("s_wait_dscnt 0" ::: "memory");
            half16 pa;
            const float* prow = pbufW + l16 * 32;
#pragma unroll
            for (int i = 0; i < 8; ++i) {
                pa[i]     = (_Float16)prow[lh * 8 + i];       // K = 0-7 / 8-15
                pa[8 + i] = (_Float16)prow[16 + lh * 8 + i];  // K = 16-23 / 24-31
            }
            asm volatile("s_wait_dscnt 0" ::: "memory");

            // ---- O += P @ V ----------------------------------------------
#pragma unroll
            for (int nt = 0; nt < 4; ++nt) {
                half16 bv = lds_frag16x32(Vtile, nt * 16, 0, VT_PITCH);
                o[nt] = __builtin_amdgcn_wmma_f32_16x16x32_f16(
                    false, pa, false, bv, (short)0, o[nt], false, false);
            }
        }
        __syncthreads();   // reads done before next issue overwrites buffer
    }

    // ---- normalize + store context as f16 [B*S, E] ------------------------
    float inv[8];
#pragma unroll
    for (int v = 0; v < 8; ++v) inv[v] = lrow[v] > 0.f ? 1.0f / lrow[v] : 0.f;
#pragma unroll
    for (int nt = 0; nt < 4; ++nt) {
        const int col = h * HDIM + nt * 16 + l16;
#pragma unroll
        for (int v = 0; v < 8; ++v) {
            const int s = qBase + lh * 8 + v;
            Ctx[(size_t)(b * SEQ + s) * EMBED + col] = (_Float16)(o[nt][v] * inv[v]);
        }
    }
}

// ---------------------------------------------------------------------------
extern "C" void kernel_launch(void* const* d_in, const int* in_sizes, int n_in,
                              void* d_out, int out_size, void* d_ws, size_t ws_size,
                              hipStream_t stream) {
    const float* X   = (const float*)d_in[0];
    const int*   Am  = (const int*)  d_in[1];
    const float* Wq  = (const float*)d_in[2];
    const float* bq  = (const float*)d_in[3];
    const float* Wk  = (const float*)d_in[4];
    const float* bk  = (const float*)d_in[5];
    const float* Wv  = (const float*)d_in[6];
    const float* bv  = (const float*)d_in[7];
    const float* Wo  = (const float*)d_in[8];
    const float* bo  = (const float*)d_in[9];
    float* out = (float*)d_out;

    // Workspace carve-up (all f16): 48 MiB total
    _Float16* ws  = (_Float16*)d_ws;
    _Float16* Xh   = ws;                              // 4096*1024
    _Float16* Wqh  = Xh  + (size_t)MTOT * EMBED;      // 1024*1024
    _Float16* Wkh  = Wqh + (size_t)EMBED * EMBED;
    _Float16* Wvh  = Wkh + (size_t)EMBED * EMBED;
    _Float16* Woh  = Wvh + (size_t)EMBED * EMBED;
    _Float16* Qh   = Woh + (size_t)EMBED * EMBED;     // [B,H,S,Dh]
    _Float16* Kh   = Qh  + (size_t)MTOT * EMBED;
    _Float16* Vth  = Kh  + (size_t)MTOT * EMBED;      // [B,H,Dh,S]
    _Float16* Ctxh = Vth + (size_t)MTOT * EMBED;      // [B*S, E]

    const int nX = MTOT * EMBED;      // 4M
    const int nW = EMBED * EMBED;     // 1M
    cvt_f32_to_f16_kernel<<<(nX + 255) / 256, 256, 0, stream>>>(X,  Xh,  nX);
    cvt_f32_to_f16_kernel<<<(nW + 255) / 256, 256, 0, stream>>>(Wq, Wqh, nW);
    cvt_f32_to_f16_kernel<<<(nW + 255) / 256, 256, 0, stream>>>(Wk, Wkh, nW);
    cvt_f32_to_f16_kernel<<<(nW + 255) / 256, 256, 0, stream>>>(Wv, Wvh, nW);
    cvt_f32_to_f16_kernel<<<(nW + 255) / 256, 256, 0, stream>>>(Wo, Woh, nW);

    dim3 gGemm(MTOT / 128, EMBED / 128);   // 32 x 8
    gemm_wmma_bias_kernel<<<gGemm, 128, GEMM_SMEM, stream>>>(Xh, Wqh, bq, Qh,
                                                             MTOT, EMBED, EMBED, 0);
    gemm_wmma_bias_kernel<<<gGemm, 128, GEMM_SMEM, stream>>>(Xh, Wkh, bk, Kh,
                                                             MTOT, EMBED, EMBED, 0);
    gemm_wmma_bias_kernel<<<gGemm, 128, GEMM_SMEM, stream>>>(Xh, Wvh, bv, Vth,
                                                             MTOT, EMBED, EMBED, 1);

    dim3 gAttn(SEQ / 64, BATCH * HEADS);
    flash_attn_wmma_kernel<<<gAttn, 128, ATTN_SMEM, stream>>>(Qh, Kh, Vth, Am, Ctxh);

    gemm_wmma_bias_kernel<<<gGemm, 128, GEMM_SMEM, stream>>>(Ctxh, Woh, bo, out,
                                                             MTOT, EMBED, EMBED, 2);
}
